// MHGCN_13288628813898
// MI455X (gfx1250) — matrix operand order repaired and checked by reading, not compiled, not run on hardware
//
#include <hip/hip_runtime.h>
#include <hip/hip_bf16.h>
#include <math.h>

#define NN    50000
#define NF    256
#define NH    128
#define NO    64
#define NEDGE 800000
#define NPL   3

typedef __attribute__((ext_vector_type(16))) _Float16 v16h;
typedef __attribute__((ext_vector_type(8)))  _Float16 v8h;
typedef __attribute__((ext_vector_type(8)))  float    v8f;

// --------------------------------------------------------------------------
// WMMA GEMM: C[M x Nc] = A[M x K] @ W[K x Nc] (+bias)(+relu)
// A, W row-major fp32; f16 multiply, fp32 accumulate.
// Per block (128 thr = 4 waves): the 64-column strip of W is staged ONCE into
// LDS, transposed to [col][K] and pre-converted to f16 (padded row stride to
// avoid bank conflicts). Each wave computes a 16x64 output strip: the 16x32 A
// fragment is built once per k-step (4x global_load_b128) and reused across 4
// v_wmma_f32_16x16x32_f16 issues; each B fragment is 2x ds_load_b128.
// REQUIRES: M % 16 == 0, K % 32 == 0, Nc % 64 == 0 (true for all call sites).
// --------------------------------------------------------------------------
#define LDS_PAD 8                        // halves; row stride (K+8)*2B = 132 dw
#define LDS_MAXK 256

template<int ACT>  // 0: none, 1: +bias, 2: +bias+relu
__global__ __launch_bounds__(128)
void wmma_gemm_kernel(const float* __restrict__ A, const float* __restrict__ W,
                      const float* __restrict__ bias, float* __restrict__ C,
                      int M, int K, int Nc) {
  __shared__ _Float16 lds[64 * (LDS_MAXK + LDS_PAD)];

  const int tid   = threadIdx.x;
  const int tileN = blockIdx.y;           // strip of 64 columns
  const int n0    = tileN * 64;
  const int ldk   = K + LDS_PAD;

  // ---- stage W strip: lds[nloc*ldk + k] = (f16) W[k*Nc + n0 + nloc] ----
  for (int idx = tid; idx < (K << 6); idx += 128) {
    const int k    = idx >> 6;
    const int nloc = idx & 63;
    lds[nloc * ldk + k] = (_Float16)W[(size_t)k * Nc + n0 + nloc];
  }
  __syncthreads();

  const int wave  = tid >> 5;
  const int lane  = tid & 31;
  const int tileM = blockIdx.x * 4 + wave;
  if (tileM * 16 >= M) return;            // wave-uniform exit (after the sync)

  const int half = lane >> 4;             // 0: lanes 0-15, 1: lanes 16-31
  const int l15  = lane & 15;
  const int mA   = tileM * 16 + l15;      // A row owned by this lane (always < M)
  const int kbA  = half * 8;
  const int kbB  = half * 8;
  const float* Arow = A + (size_t)mA * K;

  v8f acc0 = {}, acc1 = {}, acc2 = {}, acc3 = {};

  for (int k0 = 0; k0 < K; k0 += 32) {
    // A fragment (16x32 f16, ISA 7.12.2): VGPR j holds K=kb+2j,+1 ; VGPR 4+j holds K=16+kb+2j,+1
    const float4* Ap = (const float4*)(Arow + k0 + kbA);        // K = kb .. kb+7
    const float4* Aq = (const float4*)(Arow + k0 + 16 + kbA);   // K = 16+kb .. 16+kb+7
    float4 p0 = Ap[0], p1 = Ap[1];
    float4 q0 = Aq[0], q1 = Aq[1];
    v16h a;
    a[0]  = (_Float16)p0.x; a[1]  = (_Float16)p0.y;
    a[2]  = (_Float16)p0.z; a[3]  = (_Float16)p0.w;
    a[4]  = (_Float16)p1.x; a[5]  = (_Float16)p1.y;
    a[6]  = (_Float16)p1.z; a[7]  = (_Float16)p1.w;
    a[8]  = (_Float16)q0.x; a[9]  = (_Float16)q0.y;
    a[10] = (_Float16)q0.z; a[11] = (_Float16)q0.w;
    a[12] = (_Float16)q1.x; a[13] = (_Float16)q1.y;
    a[14] = (_Float16)q1.z; a[15] = (_Float16)q1.w;

    // 4 B fragments straight from LDS (K = kb..kb+7 and 16+kb..16+kb+7, same as A), A reused
#pragma unroll
    for (int t = 0; t < 4; ++t) {
      const _Float16* bp = lds + (t * 16 + l15) * ldk + k0 + kbB;
      v8h blo = *(const v8h*)(bp);
      v8h bhi = *(const v8h*)(bp + 16);
      v16h b = __builtin_shufflevector(blo, bhi, 0, 1, 2, 3, 4, 5, 6, 7,
                                       8, 9, 10, 11, 12, 13, 14, 15);
      if      (t == 0) acc0 = __builtin_amdgcn_wmma_f32_16x16x32_f16(false, a, false, b, (short)0, acc0, false, false);
      else if (t == 1) acc1 = __builtin_amdgcn_wmma_f32_16x16x32_f16(false, a, false, b, (short)0, acc1, false, false);
      else if (t == 2) acc2 = __builtin_amdgcn_wmma_f32_16x16x32_f16(false, a, false, b, (short)0, acc2, false, false);
      else             acc3 = __builtin_amdgcn_wmma_f32_16x16x32_f16(false, a, false, b, (short)0, acc3, false, false);
    }
  }

#pragma unroll
  for (int t = 0; t < 4; ++t) {
    const int nt = n0 + t * 16 + l15;
    const float bv = (ACT >= 1) ? bias[nt] : 0.0f;
    const v8f& acc = (t == 0) ? acc0 : (t == 1) ? acc1 : (t == 2) ? acc2 : acc3;
#pragma unroll
    for (int r = 0; r < 8; ++r) {
      const int row = tileM * 16 + half * 8 + r;   // C/D layout: VGPR r -> rows r / r+8
      float v = acc[r] + bv;
      if (ACT == 2) v = fmaxf(v, 0.0f);
      C[(size_t)row * Nc + nt] = v;
    }
  }
}

// --------------------------------------------------------------------------
// SPMM scatter:  Y[row[e], :] += s * X[col[e], :]   (s = val[e] * optional scale)
// One thread per (edge, 4-column group); fp32 global atomics.
// --------------------------------------------------------------------------
__global__ void spmm_kernel(const int* __restrict__ rows, const int* __restrict__ cols,
                            const float* __restrict__ vals, const float* __restrict__ X,
                            float* __restrict__ Y, int C,
                            const float* __restrict__ scale, int sidx) {
  const int e = blockIdx.x * blockDim.x + threadIdx.x;
  if (e >= NEDGE) return;
  const int g = blockIdx.y;               // column group of 4
  float s = vals[e];
  if (scale) s *= scale[sidx];
  const int r = rows[e];
  const int c = cols[e];
  const float* x = X + (size_t)c * C + g * 4;
  float*       y = Y + (size_t)r * C + g * 4;
  atomicAdd(y + 0, s * x[0]);
  atomicAdd(y + 1, s * x[1]);
  atomicAdd(y + 2, s * x[2]);
  atomicAdd(y + 3, s * x[3]);
}

__global__ void fill_kernel(float* __restrict__ p, long long n, float v) {
  long long i = (long long)blockIdx.x * 256 + threadIdx.x;
  if (i < n) p[i] = v;
}

// buf[n,c] = act(buf[n,c] + bias[c]);  TELU: x * tanh(exp(x))
template<int TELU>
__global__ void bias_act_kernel(float* __restrict__ buf, const float* __restrict__ bias,
                                int M, int C) {
  long long i = (long long)blockIdx.x * 256 + threadIdx.x;
  if (i >= (long long)M * C) return;
  int c = (int)(i % C);
  float x = buf[i] + bias[c];
  if (TELU) x = x * tanhf(expf(x));
  buf[i] = x;
}

// cat[n, p*64+c] = spec[p][n][c]   (N x 192, row-major)
__global__ void concat_spec_kernel(const float* __restrict__ spec, float* __restrict__ cat) {
  long long i = (long long)blockIdx.x * 256 + threadIdx.x;
  if (i >= (long long)NN * (NPL * NO)) return;
  int col = (int)(i % (NPL * NO));
  long long n = i / (NPL * NO);
  int p = col >> 6, c = col & 63;
  cat[i] = spec[(size_t)p * NN * NO + (size_t)n * NO + c];
}

// Per (plane, column): mean over N, max over N, entropy of softmax over N.
__global__ void col_summary_kernel(const float* __restrict__ spec, float* __restrict__ ps) {
  const int c = blockIdx.x;               // 0..63
  const int p = blockIdx.y;               // 0..2
  const float* s = spec + (size_t)p * NN * NO;
  const int tid = threadIdx.x;
  __shared__ float sh[256];
  __shared__ float sh2[256];

  float lsum = 0.0f, lmax = -3.4e38f;
  for (int n = tid; n < NN; n += 256) {
    float v = s[(size_t)n * NO + c];
    lsum += v; lmax = fmaxf(lmax, v);
  }
  sh[tid] = lsum; sh2[tid] = lmax; __syncthreads();
  for (int st = 128; st > 0; st >>= 1) {
    if (tid < st) { sh[tid] += sh[tid + st]; sh2[tid] = fmaxf(sh2[tid], sh2[tid + st]); }
    __syncthreads();
  }
  const float total = sh[0];
  const float m     = sh2[0];
  __syncthreads();

  float lz = 0.0f;
  for (int n = tid; n < NN; n += 256) lz += expf(s[(size_t)n * NO + c] - m);
  sh[tid] = lz; __syncthreads();
  for (int st = 128; st > 0; st >>= 1) { if (tid < st) sh[tid] += sh[tid + st]; __syncthreads(); }
  const float Z = sh[0];
  __syncthreads();

  float le = 0.0f;
  for (int n = tid; n < NN; n += 256) {
    float pr = expf(s[(size_t)n * NO + c] - m) / Z;
    le += pr * logf(pr + 1e-6f);
  }
  sh[tid] = le; __syncthreads();
  for (int st = 128; st > 0; st >>= 1) { if (tid < st) sh[tid] += sh[tid + st]; __syncthreads(); }

  if (tid == 0) {
    ps[p * 192 + c]       = total / (float)NN;   // mean
    ps[p * 192 + 64 + c]  = m;                   // max
    ps[p * 192 + 128 + c] = -sh[0];              // entropy
  }
}

// 3x3 similarity softmax + 13-step random walk; emits w (=pi) and softmax(pi).
__global__ void walk_kernel(const float* __restrict__ ps, const float* __restrict__ wb,
                            const float* __restrict__ tau,
                            float* __restrict__ wvec, float* __restrict__ wprob) {
  if (threadIdx.x != 0 || blockIdx.x != 0) return;
  float sim[3][3];
  const float inv = 1.0f / (sqrtf(192.0f) * tau[0]);
  for (int i = 0; i < 3; ++i)
    for (int j = 0; j < 3; ++j) {
      float d = 0.0f;
      for (int k = 0; k < 192; ++k) d += ps[i * 192 + k] * ps[j * 192 + k];
      sim[i][j] = d * inv;
    }
  float T[3][3];
  for (int i = 0; i < 3; ++i) {
    float m = fmaxf(sim[i][0], fmaxf(sim[i][1], sim[i][2]));
    float z = 0.0f;
    for (int j = 0; j < 3; ++j) { T[i][j] = expf(sim[i][j] - m); z += T[i][j]; }
    for (int j = 0; j < 3; ++j) T[i][j] /= z;
  }
  float p0[3];
  { float m = fmaxf(wb[0], fmaxf(wb[1], wb[2])); float z = 0.0f;
    for (int i = 0; i < 3; ++i) { p0[i] = expf(wb[i] - m); z += p0[i]; }
    for (int i = 0; i < 3; ++i) p0[i] /= z; }
  float pi[3] = { p0[0], p0[1], p0[2] };
  for (int s = 0; s < 13; ++s) {
    float np[3];
    for (int j = 0; j < 3; ++j) {
      float a = 0.0f;
      for (int i = 0; i < 3; ++i) a += pi[i] * T[i][j];
      np[j] = 0.2f * p0[j] + 0.8f * a;
    }
    pi[0] = np[0]; pi[1] = np[1]; pi[2] = np[2];
  }
  for (int i = 0; i < 3; ++i) wvec[i] = pi[i];
  { float m = fmaxf(pi[0], fmaxf(pi[1], pi[2])); float z = 0.0f; float e[3];
    for (int i = 0; i < 3; ++i) { e[i] = expf(pi[i] - m); z += e[i]; }
    for (int i = 0; i < 3; ++i) wprob[i] = e[i] / z; }
}

__global__ void hraw_kernel(const float* __restrict__ U1, const float* __restrict__ U2,
                            float* __restrict__ hraw) {
  long long i = (long long)blockIdx.x * 256 + threadIdx.x;
  if (i >= (long long)NN * NO) return;
  hraw[i] = 0.5f * (U1[i] + U2[i]);
}

// all_feat[n,0:64]=Σ wprob[p]*spec ; [64:128]=mean(shared) ; [128:192]=H_col ; [192:256]=H_raw
__global__ void assemble_kernel(const float* __restrict__ spec, const float* __restrict__ shrd,
                                const float* __restrict__ hcol, const float* __restrict__ hraw,
                                const float* __restrict__ wprob, float* __restrict__ allfeat) {
  long long i = (long long)blockIdx.x * 256 + threadIdx.x;
  if (i >= (long long)NN * NO) return;
  long long n = i >> 6;
  int c = (int)(i & 63);
  size_t off = (size_t)n * NO + c;
  const size_t PL = (size_t)NN * NO;
  float fsp = wprob[0] * spec[off] + wprob[1] * spec[PL + off] + wprob[2] * spec[2 * PL + off];
  float msh = (shrd[off] + shrd[PL + off] + shrd[2 * PL + off]) * (1.0f / 3.0f);
  float* row = allfeat + (size_t)n * 256;
  row[c]       = fsp;
  row[64 + c]  = msh;
  row[128 + c] = hcol[off];
  row[192 + c] = hraw[off];
}

// --------------------------------------------------------------------------
extern "C" void kernel_launch(void* const* d_in, const int* in_sizes, int n_in,
                              void* d_out_v, int out_size, void* d_ws, size_t ws_size,
                              hipStream_t stream) {
  (void)in_sizes; (void)n_in; (void)out_size; (void)ws_size;
  const float* feature = (const float*)d_in[0];
  const int*   eidx    = (const int*)  d_in[1];   // (P,2,E)
  const float* evalp   = (const float*)d_in[2];   // (P,E)
  const float* spec_w1 = (const float*)d_in[3];
  const float* spec_b1 = (const float*)d_in[4];
  const float* spec_w2 = (const float*)d_in[5];
  const float* spec_b2 = (const float*)d_in[6];
  const float* sh_w1   = (const float*)d_in[7];
  const float* sh_b1   = (const float*)d_in[8];
  const float* sh_w2   = (const float*)d_in[9];
  const float* sh_b2   = (const float*)d_in[10];
  const float* wb      = (const float*)d_in[11];
  const float* tau     = (const float*)d_in[12];
  const float* mlp_w1  = (const float*)d_in[13];
  const float* mlp_b1  = (const float*)d_in[14];
  const float* mlp_w2  = (const float*)d_in[15];
  const float* mlp_b2  = (const float*)d_in[16];
  const float* raw_w1  = (const float*)d_in[17];
  const float* raw_b1  = (const float*)d_in[18];
  const float* raw_w2  = (const float*)d_in[19];
  const float* raw_b2  = (const float*)d_in[20];
  const float* proj_w  = (const float*)d_in[21];
  const float* proj_b  = (const float*)d_in[22];

  // d_out layout: out | spec_stack | shared_stack | H_col | H_raw
  float* outp   = (float*)d_out_v;
  float* spec_o = outp + (size_t)NN * NO;
  float* shrd_o = spec_o + (size_t)NPL * NN * NO;
  float* hcol   = shrd_o + (size_t)NPL * NN * NO;
  float* hraw   = hcol + (size_t)NN * NO;

  // workspace: allfeat(N*256, reused as tmp64/concat) | t0(N*128) | t1(N*128) | ps(576) | w(3) | wprob(3)
  float* allfeat = (float*)d_ws;
  float* t0 = allfeat + (size_t)NN * 256;
  float* t1 = t0 + (size_t)NN * NH;
  float* ps = t1 + (size_t)NN * NH;
  float* wvec  = ps + 576;
  float* wprob = wvec + 3;

  auto gemm = [&](const float* A, const float* W, const float* b, float* Cm,
                  int K, int Nc, int act) {
    dim3 grid((NN / 16 + 3) / 4, Nc / 64), blk(128);
    if (act == 0)
      hipLaunchKernelGGL((wmma_gemm_kernel<0>), grid, blk, 0, stream, A, W, b, Cm, NN, K, Nc);
    else if (act == 1)
      hipLaunchKernelGGL((wmma_gemm_kernel<1>), grid, blk, 0, stream, A, W, b, Cm, NN, K, Nc);
    else
      hipLaunchKernelGGL((wmma_gemm_kernel<2>), grid, blk, 0, stream, A, W, b, Cm, NN, K, Nc);
  };
  auto fill = [&](float* p, long long n) {
    hipLaunchKernelGGL(fill_kernel, dim3((unsigned)((n + 255) / 256)), dim3(256), 0, stream,
                       p, n, 0.0f);
  };
  auto spmm = [&](int plane, const float* X, float* Y, int C, const float* scale, int sidx) {
    const int* rows = eidx + (size_t)plane * 2 * NEDGE;
    const int* cols = rows + NEDGE;
    const float* vals = evalp + (size_t)plane * NEDGE;
    dim3 grid((NEDGE + 255) / 256, C / 4);
    hipLaunchKernelGGL(spmm_kernel, grid, dim3(256), 0, stream, rows, cols, vals, X, Y, C,
                       scale, sidx);
  };
  auto biasact = [&](float* buf, const float* b, int C, int telu) {
    long long n = (long long)NN * C;
    dim3 grid((unsigned)((n + 255) / 256));
    if (telu) hipLaunchKernelGGL(bias_act_kernel<1>, grid, dim3(256), 0, stream, buf, b, NN, C);
    else      hipLaunchKernelGGL(bias_act_kernel<0>, grid, dim3(256), 0, stream, buf, b, NN, C);
  };

  // 1) plane-specific branches
  for (int i = 0; i < NPL; ++i) {
    float* so = spec_o + (size_t)i * NN * NO;
    gemm(feature, spec_w1 + (size_t)i * NF * NH, nullptr, t0, NF, NH, 0);
    fill(t1, (long long)NN * NH);
    spmm(i, t0, t1, NH, nullptr, 0);
    biasact(t1, spec_b1 + (size_t)i * NH, NH, 1);                 // TeLU
    gemm(t1, spec_w2 + (size_t)i * NH * NO, nullptr, t0, NH, NO, 0);
    fill(so, (long long)NN * NO);
    spmm(i, t0, so, NO, nullptr, 0);
    biasact(so, spec_b2 + (size_t)i * NO, NO, 0);
  }
  // 2) shared branches (feature @ shared_w1 hoisted across planes)
  gemm(feature, sh_w1, nullptr, t0, NF, NH, 0);
  for (int i = 0; i < NPL; ++i) {
    float* so = shrd_o + (size_t)i * NN * NO;
    fill(t1, (long long)NN * NH);
    spmm(i, t0, t1, NH, nullptr, 0);
    biasact(t1, sh_b1, NH, 1);                                    // TeLU
    gemm(t1, sh_w2, nullptr, allfeat /* tmp N x 64 */, NH, NO, 0);
    fill(so, (long long)NN * NO);
    spmm(i, allfeat, so, NO, nullptr, 0);
    biasact(so, sh_b2, NO, 0);
  }
  // 3) plane summaries + random walk (device-side, deterministic)
  hipLaunchKernelGGL(col_summary_kernel, dim3(NO, NPL), dim3(256), 0, stream, spec_o, ps);
  hipLaunchKernelGGL(walk_kernel, dim3(1), dim3(32), 0, stream, ps, wb, tau, wvec, wprob);
  // 4) H_col = relu(concat_sp @ mlp_w1 + b1) @ mlp_w2 + b2
  hipLaunchKernelGGL(concat_spec_kernel,
                     dim3((unsigned)(((long long)NN * NPL * NO + 255) / 256)), dim3(256), 0,
                     stream, spec_o, allfeat);
  gemm(allfeat, mlp_w1, mlp_b1, t0, NPL * NO, NH, 2);             // bias+relu
  gemm(t0, mlp_w2, mlp_b2, hcol, NH, NO, 1);                      // bias
  // 5) H_raw via pi-weighted merged SPMM
  gemm(feature, raw_w1, nullptr, t0, NF, NO, 0);                  // Y = feature @ raw_w1
  fill(t1, (long long)NN * NO);
  for (int i = 0; i < NPL; ++i) spmm(i, t0, t1, NO, wvec, i);
  biasact(t1, raw_b1, NO, 0);                                     // U1 = t1
  gemm(t1, raw_w2, nullptr, t0, NO, NO, 0);                       // Y2 overwrites Y
  float* u2 = t0 + (size_t)NN * NO;
  fill(u2, (long long)NN * NO);
  for (int i = 0; i < NPL; ++i) spmm(i, t0, u2, NO, wvec, i);
  biasact(u2, raw_b2, NO, 0);                                     // U2
  hipLaunchKernelGGL(hraw_kernel, dim3((unsigned)(((long long)NN * NO + 255) / 256)),
                     dim3(256), 0, stream, t1, u2, hraw);
  // 6) assemble all_feat (N x 256) and final projection
  hipLaunchKernelGGL(assemble_kernel, dim3((unsigned)(((long long)NN * NO + 255) / 256)),
                     dim3(256), 0, stream, spec_o, shrd_o, hcol, hraw, wprob, allfeat);
  gemm(allfeat, proj_w, proj_b, outp, 256, NO, 1);
}
